// PytorchFusedJaggedBmmSwigluCombineModule_56332791054818
// MI455X (gfx1250) — matrix operand order repaired and checked by reading, not compiled, not run on hardware
//
#include <hip/hip_runtime.h>

// Problem constants (fixed by the reference)
#define TOK 8192
#define DD  1024
#define HH  1024
#define EE  8
#define KK  2
#define NN  (TOK * KK)

// moe tile config: 32 rows per block, 512 threads = 16 waves,
// each wave owns 64 columns (4 16x16 tiles) and both 16-row subtiles.
#define MROWS 32
#define HSPITCH (HH + 8)   // padded LDS row (bf16 elements)

typedef __bf16 bf16;
typedef __attribute__((ext_vector_type(4)))  bf16  v4bf;   //  8 B
typedef __attribute__((ext_vector_type(8)))  bf16  v8bf;   // 16 B
typedef __attribute__((ext_vector_type(16))) bf16  v16bf;  // 32 B (8 VGPRs)
typedef __attribute__((ext_vector_type(8)))  float v8f;    // WMMA f32 accumulator

static __device__ __forceinline__ v16bf cat16(v8bf lo, v8bf hi) {
  v16bf r;
#pragma unroll
  for (int i = 0; i < 8; ++i) { r[i] = lo[i]; r[8 + i] = hi[i]; }
  return r;
}

static __device__ __forceinline__ v8f vzero8() {
  v8f r;
#pragma unroll
  for (int i = 0; i < 8; ++i) r[i] = 0.0f;
  return r;
}

// ---------------------------------------------------------------------------
// Prep 1: x = bf16(jagged[index])   [N, D]
// ---------------------------------------------------------------------------
__global__ __launch_bounds__(256) void gather_cast_x(
    const float* __restrict__ jagged, const int* __restrict__ index,
    bf16* __restrict__ xbf) {
  const int n = blockIdx.x;
  const int t = index[n];
  const int d = threadIdx.x * 4;
  const float4 v = *(const float4*)(jagged + (size_t)t * DD + d);
  v4bf o;
  o[0] = (bf16)v.x; o[1] = (bf16)v.y; o[2] = (bf16)v.z; o[3] = (bf16)v.w;
  *(v4bf*)(xbf + (size_t)n * DD + d) = o;
}

// ---------------------------------------------------------------------------
// Prep 2: transpose + cast f32 [E][R][C] -> bf16 [E][C][R]
// Makes the WMMA B operand K-contiguous in memory.
// ---------------------------------------------------------------------------
__global__ __launch_bounds__(256) void transpose_cvt(
    const float* __restrict__ src, bf16* __restrict__ dst, int R, int C) {
  __shared__ float tile[32][33];
  const int e  = blockIdx.z;
  const int cb = blockIdx.x * 32;
  const int rb = blockIdx.y * 32;
  const float* s = src + (size_t)e * R * C;
  bf16*        d = dst + (size_t)e * R * C;
#pragma unroll
  for (int i = 0; i < 4; ++i) {
    const int r = rb + threadIdx.y + i * 8;
    tile[threadIdx.y + i * 8][threadIdx.x] = s[(size_t)r * C + cb + threadIdx.x];
  }
  __syncthreads();
#pragma unroll
  for (int i = 0; i < 4; ++i) {
    const int c = cb + threadIdx.y + i * 8;
    d[(size_t)c * R + rb + threadIdx.x] = (bf16)tile[threadIdx.x][threadIdx.y + i * 8];
  }
}

// ---------------------------------------------------------------------------
// Main fused MoE FFN, 32-row tile per block:
//   phase 1: g = xW+b, u = xWp+bp, h = silu(g)*u  -> LDS (bf16)
//   phase 2: y = h Wout + bout                    -> global (f32)
// Fragment addressing (16-bit WMMA operands, wave32):
//   A (16xK row-major):  lane m=lane&15, half=lane>>4:
//       [m][k0+half*16 .. +15]                              (2x b128)
//   B (Kx16 from K-contiguous transposed weights wT[n][k]): lane n=lane&15:
//       wT[n][k0+half*16 .. +15]                            (2x b128)
// B fragments are reused across both 16-row subtiles (2 WMMAs per B load).
// ---------------------------------------------------------------------------
__global__ __launch_bounds__(512) void moe_ffn_wmma(
    const bf16* __restrict__ xbf,   // [N][D]
    const bf16* __restrict__ w1T,   // [E][H][D]
    const bf16* __restrict__ wpT,   // [E][H][D]
    const bf16* __restrict__ woT,   // [E][D][H]
    const float* __restrict__ bias,    // [E][H]
    const float* __restrict__ bias_p,  // [E][H]
    const float* __restrict__ bias_o,  // [E][D]
    const int* __restrict__ offsets,   // [E+1]
    float* __restrict__ y) {           // [N][D]
  extern __shared__ bf16 hs[];         // [MROWS][HSPITCH]

  const int row0 = blockIdx.x * MROWS;
  int e = 0;
  while (e + 1 < EE && offsets[e + 1] <= row0) ++e;

  const int lane = threadIdx.x & 31;
  const int wave = threadIdx.x >> 5;
  const int half = lane >> 4;  // K-halves of the fragment
  const int ln   = lane & 15;  // M (A) or N (B) index

  const bf16* W1 = w1T + (size_t)e * HH * DD;
  const bf16* WP = wpT + (size_t)e * HH * DD;
  const bf16* WO = woT + (size_t)e * DD * HH;

  // ---------------- phase 1: g, u, SwiGLU -> hs ----------------
  v8f accg[2][4], accu[2][4];
#pragma unroll
  for (int i = 0; i < 2; ++i)
#pragma unroll
    for (int c = 0; c < 4; ++c) { accg[i][c] = vzero8(); accu[i][c] = vzero8(); }

  const int wbase = wave * 64;
  for (int k0 = 0; k0 < DD; k0 += 32) {
    v16bf a[2];
#pragma unroll
    for (int i = 0; i < 2; ++i) {
      const bf16* arow =
          xbf + (size_t)(row0 + i * 16 + ln) * DD + k0 + half * 16;
      a[i] = cat16(*(const v8bf*)arow, *(const v8bf*)(arow + 8));
    }
#pragma unroll
    for (int c = 0; c < 4; ++c) {
      const bf16* b1 = W1 + (size_t)(wbase + c * 16 + ln) * DD + k0 + half * 16;
      const v16bf bg = cat16(*(const v8bf*)b1, *(const v8bf*)(b1 + 8));
#pragma unroll
      for (int i = 0; i < 2; ++i)
        accg[i][c] = __builtin_amdgcn_wmma_f32_16x16x32_bf16(
            false, a[i], false, bg, (short)0, accg[i][c], false, false);
      const bf16* b2 = WP + (size_t)(wbase + c * 16 + ln) * DD + k0 + half * 16;
      const v16bf bu = cat16(*(const v8bf*)b2, *(const v8bf*)(b2 + 8));
#pragma unroll
      for (int i = 0; i < 2; ++i)
        accu[i][c] = __builtin_amdgcn_wmma_f32_16x16x32_bf16(
            false, a[i], false, bu, (short)0, accu[i][c], false, false);
    }
  }

  // epilogue: bias + SiLU, store bf16 h-tile to LDS
  // C/D layout: element (M = v + 8*half, N = ln) lives in acc[v]
#pragma unroll
  for (int c = 0; c < 4; ++c) {
    const int col = wbase + c * 16 + ln;
    const float bg = bias[(size_t)e * HH + col];
    const float bu = bias_p[(size_t)e * HH + col];
#pragma unroll
    for (int i = 0; i < 2; ++i) {
#pragma unroll
      for (int v = 0; v < 8; ++v) {
        const int m = i * 16 + v + half * 8;
        const float g = accg[i][c][v] + bg;
        const float u = accu[i][c][v] + bu;
        const float s = g / (1.0f + __expf(-g));  // silu(g)
        hs[(size_t)m * HSPITCH + col] = (bf16)(s * u);
      }
    }
  }
  __syncthreads();

  // ---------------- phase 2: y = h @ Wout + bias_out ----------------
  v8f acc[2][4];
#pragma unroll
  for (int i = 0; i < 2; ++i)
#pragma unroll
    for (int c = 0; c < 4; ++c) acc[i][c] = vzero8();

  const int dbase = wave * 64;
  for (int k0 = 0; k0 < HH; k0 += 32) {
    v16bf a[2];
#pragma unroll
    for (int i = 0; i < 2; ++i) {
      const bf16* ap = hs + (size_t)(i * 16 + ln) * HSPITCH + k0 + half * 16;
      a[i] = cat16(*(const v8bf*)ap, *(const v8bf*)(ap + 8));
    }
#pragma unroll
    for (int c = 0; c < 4; ++c) {
      const bf16* bp = WO + (size_t)(dbase + c * 16 + ln) * HH + k0 + half * 16;
      const v16bf b = cat16(*(const v8bf*)bp, *(const v8bf*)(bp + 8));
#pragma unroll
      for (int i = 0; i < 2; ++i)
        acc[i][c] = __builtin_amdgcn_wmma_f32_16x16x32_bf16(
            false, a[i], false, b, (short)0, acc[i][c], false, false);
    }
  }

#pragma unroll
  for (int c = 0; c < 4; ++c) {
    const int col = dbase + c * 16 + ln;
    const float bo = bias_o[(size_t)e * DD + col];
#pragma unroll
    for (int i = 0; i < 2; ++i) {
#pragma unroll
      for (int v = 0; v < 8; ++v) {
        const int m = i * 16 + v + half * 8;
        y[(size_t)(row0 + m) * DD + col] = acc[i][c][v] + bo;
      }
    }
  }
}

// ---------------------------------------------------------------------------
// Combine: out[t] = sum_k y[rev[t,k]] * gates[gidx[t,k]]
// ---------------------------------------------------------------------------
__global__ __launch_bounds__(256) void combine_kernel(
    const float* __restrict__ y, const int* __restrict__ rev,
    const int* __restrict__ gidx, const float* __restrict__ gates,
    float* __restrict__ out) {
  const int t = blockIdx.x;
  const int r0 = rev[t * KK + 0];
  const int r1 = rev[t * KK + 1];
  const float g0 = gates[gidx[t * KK + 0]];
  const float g1 = gates[gidx[t * KK + 1]];
  const int d = threadIdx.x * 4;
  const float4 a = *(const float4*)(y + (size_t)r0 * DD + d);
  const float4 b = *(const float4*)(y + (size_t)r1 * DD + d);
  float4 o;
  o.x = a.x * g0 + b.x * g1;
  o.y = a.y * g0 + b.y * g1;
  o.z = a.z * g0 + b.z * g1;
  o.w = a.w * g0 + b.w * g1;
  *(float4*)(out + (size_t)t * DD + d) = o;
}

// ---------------------------------------------------------------------------
extern "C" void kernel_launch(void* const* d_in, const int* in_sizes, int n_in,
                              void* d_out, int out_size, void* d_ws,
                              size_t ws_size, hipStream_t stream) {
  (void)in_sizes; (void)n_in; (void)out_size; (void)ws_size;

  const int*   offsets    = (const int*)d_in[0];
  const float* jagged     = (const float*)d_in[1];
  const float* weight     = (const float*)d_in[2];
  const float* bias       = (const float*)d_in[3];
  const int*   index      = (const int*)d_in[4];
  const float* weight_p   = (const float*)d_in[5];
  const float* weight_out = (const float*)d_in[6];
  const int*   rev        = (const int*)d_in[7];
  const float* gates      = (const float*)d_in[8];
  const int*   gidx       = (const int*)d_in[9];
  const float* bias_p     = (const float*)d_in[10];
  const float* bias_out   = (const float*)d_in[11];
  float* out = (float*)d_out;

  // Workspace layout (~144 MB total)
  char* ws = (char*)d_ws;
  bf16* xbf = (bf16*)ws;  ws += (size_t)NN * DD * sizeof(bf16);      // 32 MB
  bf16* w1T = (bf16*)ws;  ws += (size_t)EE * DD * HH * sizeof(bf16); // 16 MB
  bf16* wpT = (bf16*)ws;  ws += (size_t)EE * DD * HH * sizeof(bf16); // 16 MB
  bf16* woT = (bf16*)ws;  ws += (size_t)EE * DD * HH * sizeof(bf16); // 16 MB
  float* y  = (float*)ws;                                            // 64 MB

  gather_cast_x<<<NN, 256, 0, stream>>>(jagged, index, xbf);

  dim3 tb(32, 8);
  transpose_cvt<<<dim3(HH / 32, DD / 32, EE), tb, 0, stream>>>(weight,     w1T, DD, HH);
  transpose_cvt<<<dim3(HH / 32, DD / 32, EE), tb, 0, stream>>>(weight_p,   wpT, DD, HH);
  transpose_cvt<<<dim3(DD / 32, HH / 32, EE), tb, 0, stream>>>(weight_out, woT, HH, DD);

  const size_t hs_bytes = (size_t)MROWS * HSPITCH * sizeof(bf16);  // 66 KB
  moe_ffn_wmma<<<NN / MROWS, 512, hs_bytes, stream>>>(
      xbf, w1T, wpT, woT, bias, bias_p, bias_out, offsets, y);

  combine_kernel<<<TOK, 256, 0, stream>>>(y, rev, gidx, gates, out);
}
